// Bert_1047972020447
// MI455X (gfx1250) — compile-verified, matrix operand order checked by reading, not run on hardware
//
#include <hip/hip_runtime.h>

// ---------------------------------------------------------------------------
// BERT encoder stack for MI455X (gfx1250, wave32, WMMA).
// - All GEMMs: v_wmma_f32_16x16x32_bf16 (bf16 operands, fp32 accum).
// - Flash-fused attention (scores never hit HBM; keeps model compute-bound:
//   weights are only ~91 MB total vs 23.3 TB/s HBM).
// - LDS staging via GLOBAL_LOAD_ASYNC_TO_LDS_B128 (ASYNCcnt) where the copy
//   is transpose-free, falling back to the register path otherwise.
// - Softmax row reductions via DPP16 row ops instead of ds_bpermute.
// ---------------------------------------------------------------------------

typedef __bf16 bf16;
typedef __attribute__((ext_vector_type(16))) __bf16 bf16x16;
typedef __attribute__((ext_vector_type(8)))  __bf16 bf16x8;
typedef __attribute__((ext_vector_type(8)))  float  f32x8;
typedef int v4i __attribute__((vector_size(16)));   // builtin's b128 unit

#define NLAYER 12
#define NHEAD  12
#define EDIM   768
#define DDIM   64
#define NTOK   2048   // B*S = 4*512
#define SEQ    512

#if __has_builtin(__builtin_amdgcn_global_load_async_to_lds_b128) && \
    __has_builtin(__builtin_amdgcn_s_wait_asynccnt)
#define USE_ASYNC_LDS 1
#else
#define USE_ASYNC_LDS 0
#endif

#if __has_builtin(__builtin_amdgcn_update_dpp)
#define USE_DPP 1
#else
#define USE_DPP 0
#endif

// ---- async memory -> LDS copy of 16 bytes (per lane), ASYNCcnt tracked ----
__device__ __forceinline__ void async_copy16(const bf16* g, bf16* l) {
#if USE_ASYNC_LDS
    __builtin_amdgcn_global_load_async_to_lds_b128(
        (__attribute__((address_space(1))) v4i*)(void*)g,
        (__attribute__((address_space(3))) v4i*)(void*)l,
        /*imm offset*/0, /*cpol*/0);
#else
    *(bf16x8*)l = *(const bf16x8*)g;
#endif
}

__device__ __forceinline__ void wait_async() {
#if USE_ASYNC_LDS
    __builtin_amdgcn_s_wait_asynccnt(0);
#endif
}

// ---- 16-lane-group reductions (rows of the WMMA C layout) -----------------
// DPP16: quad_perm xor1 (0xB1), quad_perm xor2 (0x4E), row_ror:4 (0x124),
// row_ror:8 (0x128).  After xor1+xor2 each quad is uniform, so the two
// rotations complete the 16-lane reduction; DPP rows (16 lanes) keep the
// lane 0-15 / 16-31 halves independent, matching the C-tile row split.
#if USE_DPP
template <int CTRL>
__device__ __forceinline__ float dpp_movf(float v) {
    return __int_as_float(
        __builtin_amdgcn_update_dpp(0, __float_as_int(v), CTRL, 0xF, 0xF, true));
}
__device__ __forceinline__ float red_max16(float v) {
    v = fmaxf(v, dpp_movf<0xB1>(v));
    v = fmaxf(v, dpp_movf<0x4E>(v));
    v = fmaxf(v, dpp_movf<0x124>(v));
    v = fmaxf(v, dpp_movf<0x128>(v));
    return v;
}
__device__ __forceinline__ float red_sum16(float v) {
    v += dpp_movf<0xB1>(v);
    v += dpp_movf<0x4E>(v);
    v += dpp_movf<0x124>(v);
    v += dpp_movf<0x128>(v);
    return v;
}
#else
__device__ __forceinline__ float red_max16(float v) {
    v = fmaxf(v, __shfl_xor(v, 1));
    v = fmaxf(v, __shfl_xor(v, 2));
    v = fmaxf(v, __shfl_xor(v, 4));
    v = fmaxf(v, __shfl_xor(v, 8));
    return v;
}
__device__ __forceinline__ float red_sum16(float v) {
    v += __shfl_xor(v, 1);
    v += __shfl_xor(v, 2);
    v += __shfl_xor(v, 4);
    v += __shfl_xor(v, 8);
    return v;
}
#endif

// Build a 16x32 bf16 WMMA operand (A or B) for this lane from two 16-byte
// chunks.  Per the CDNA5 ISA 16-bit operand layout:
//   lanes 0-15 : elements are K = {0..7} then {16..23}
//   lanes 16-31: elements are K = {8..15} then {24..31}
__device__ __forceinline__ bf16x16 make_op(const bf16* p0, const bf16* p1) {
    union { bf16x16 v; bf16x8 h[2]; } u;
    u.h[0] = *(const bf16x8*)p0;
    u.h[1] = *(const bf16x8*)p1;
    return u.v;
}

__device__ __forceinline__ f32x8 wmma_bf16(bf16x16 a, bf16x16 b, f32x8 c) {
    return __builtin_amdgcn_wmma_f32_16x16x32_bf16(
        /*neg_a=*/false, a, /*neg_b=*/false, b,
        /*c_mod=*/(short)0, c, /*reuse_a=*/false, /*reuse_b=*/false);
}

// ---------------------------------------------------------------------------
// Generic tiled WMMA GEMM:  C[M,N] = A[M,K] @ B
//   B_IS_NK = false : B is [K,N] row-major (ldb = row stride)
//   B_IS_NK = true  : B is [N,K] row-major (torch Linear weight; A@B^T)
//   OUT_F32 : write fp32 C, else bf16 C.
// Block: 256 threads (8 waves).  Block tile 128(M) x 64(N), K-chunk 32.
// ---------------------------------------------------------------------------
template <bool B_IS_NK, bool OUT_F32>
__global__ __launch_bounds__(256)
void gemm_wmma(const bf16* __restrict__ A, const bf16* __restrict__ B,
               bf16* __restrict__ Cb, float* __restrict__ Cf,
               int M, int N, int K, int lda, int ldb, int ldc,
               long sA, long sB, long sC)
{
    __shared__ __align__(16) bf16 Al[128][32];
    __shared__ __align__(16) bf16 Bt[64][32];   // stored as [n][k]

    const int t    = threadIdx.x;
    const int lane = t & 31;
    const int w    = t >> 5;
    const int n0   = blockIdx.x * 64;
    const int m0   = blockIdx.y * 128;
    const long z   = blockIdx.z;

    A += z * sA;
    B += z * sB;

    const int hi   = (lane & 16) ? 1 : 0;
    const int l15  = lane & 15;
    const int koff = hi ? 8 : 0;

    f32x8 acc[4] = {};

    for (int k0 = 0; k0 < K; k0 += 32) {
        // ---- stage A tile [128 x 32]: pure copy -> async to LDS ----
        #pragma unroll
        for (int rep = 0; rep < 2; ++rep) {
            int id = t + rep * 256;
            int r = id >> 2, c = (id & 3) * 8;
            async_copy16(&A[(long)(m0 + r) * lda + k0 + c], &Al[r][c]);
        }
        // ---- stage B tile into [64 n][32 k] LDS layout ----
        if (B_IS_NK) {  // pure copy -> async to LDS
            int n = t >> 2, c = (t & 3) * 8;
            async_copy16(&B[(long)(n0 + n) * ldb + k0 + c], &Bt[n][c]);
        } else {        // transpose: register path
            int kc = t >> 3, c = (t & 7) * 8;
            bf16x8 v = *(const bf16x8*)&B[(long)(k0 + kc) * ldb + n0 + c];
            #pragma unroll
            for (int i = 0; i < 8; ++i) Bt[c + i][kc] = v[i];
        }
        // prefetch next K-chunk (global_prefetch_b8)
        if (k0 + 32 < K) {
            __builtin_prefetch(&A[(long)(m0 + (t >> 1)) * lda + k0 + 32], 0, 1);
            if (B_IS_NK)
                __builtin_prefetch(&B[(long)(n0 + (t >> 2)) * ldb + k0 + 32], 0, 1);
            else
                __builtin_prefetch(&B[(long)(k0 + 32 + (t >> 3)) * ldb + n0], 0, 1);
        }
        wait_async();
        __syncthreads();

        const bf16* arow = &Al[w * 16 + l15][0];
        bf16x16 a = make_op(arow + koff, arow + koff + 16);
        #pragma unroll
        for (int nt = 0; nt < 4; ++nt) {
            const bf16* brow = &Bt[nt * 16 + l15][0];
            bf16x16 b = make_op(brow + koff, brow + koff + 16);
            acc[nt] = wmma_bf16(a, b, acc[nt]);
        }
        __syncthreads();
    }

    // ---- epilogue (C layout: VGPR j -> row j (lanes<16) / j+8 (lanes>=16)) --
    const int rbase = m0 + w * 16 + hi * 8;
    #pragma unroll
    for (int nt = 0; nt < 4; ++nt) {
        #pragma unroll
        for (int j = 0; j < 8; ++j) {
            long r = rbase + j;
            long c = n0 + nt * 16 + l15;
            if (OUT_F32) Cf[z * sC + r * ldc + c] = acc[nt][j];
            else         Cb[z * sC + r * ldc + c] = (bf16)acc[nt][j];
        }
    }
}

// ---------------------------------------------------------------------------
// Flash attention (q == k == v, faithful to reference).
// Grid: (16 query-blocks, 12 heads), 256 threads (8 waves).
// Wave w owns a 16-query tile; waves share the LDS-staged 32-key block.
// Output written directly in the reference's zc = z.reshape(768, 2048) layout:
//   zc[h*64 + n/32][(n%32)*64 + d] = z[h][n][d]
// ---------------------------------------------------------------------------
__global__ __launch_bounds__(256)
void flash_attn(const bf16* __restrict__ Q, bf16* __restrict__ ZC)
{
    __shared__ __align__(16) bf16 Kb[32][64];       // [key][dim]
    __shared__ __align__(16) bf16 Kt[64][32];       // [dim][key]
    __shared__ __align__(16) bf16 Pl[8][16][32];    // per-wave P bounce

    const int t    = threadIdx.x;
    const int lane = t & 31;
    const int w    = t >> 5;
    const int h    = blockIdx.y;
    const int q0   = blockIdx.x * 128 + w * 16;
    const int hi   = (lane & 16) ? 1 : 0;
    const int l15  = lane & 15;
    const int koff = hi ? 8 : 0;

    const bf16* Qh = Q + (long)h * NTOK * DDIM;

    // Q tile as A operands for d = 0..31 and d = 32..63
    const bf16* qrow = Qh + (long)(q0 + l15) * DDIM;
    bf16x16 aq0 = make_op(qrow + koff,      qrow + koff + 16);
    bf16x16 aq1 = make_op(qrow + 32 + koff, qrow + 32 + koff + 16);

    f32x8 o[4] = {};
    float mrow[8], lrow[8];
    #pragma unroll
    for (int j = 0; j < 8; ++j) { mrow[j] = -1e30f; lrow[j] = 0.0f; }

    const int skey = t >> 3;            // staging: this thread's key row
    const int sc   = (t & 7) * 8;       // staging: this thread's dim chunk

    for (int kb = 0; kb < NTOK; kb += 32) {
        // ---- stage K/V block: async memory->LDS, then build transpose ----
        async_copy16(&Qh[(long)(kb + skey) * DDIM + sc], &Kb[skey][sc]);
        if (kb + 32 < NTOK)
            __builtin_prefetch(&Qh[(long)(kb + 32 + skey) * DDIM], 0, 1);
        wait_async();   // own 16B is now in LDS; read it back for transpose
        {
            bf16x8 v = *(const bf16x8*)&Kb[skey][sc];
            #pragma unroll
            for (int i = 0; i < 8; ++i) Kt[sc + i][skey] = v[i];
        }
        __syncthreads();

        // ---- S = Q K^T for 16 queries x 32 keys (two 16x16 C tiles) ----
        f32x8 s0 = {}, s1 = {};
        {
            const bf16* kr0 = &Kb[l15][0];        // keys  0..15 (B: N=key)
            const bf16* kr1 = &Kb[16 + l15][0];   // keys 16..31
            bf16x16 b00 = make_op(kr0 + koff,      kr0 + koff + 16);
            bf16x16 b01 = make_op(kr0 + 32 + koff, kr0 + 32 + koff + 16);
            bf16x16 b10 = make_op(kr1 + koff,      kr1 + koff + 16);
            bf16x16 b11 = make_op(kr1 + 32 + koff, kr1 + 32 + koff + 16);
            s0 = wmma_bf16(aq0, b00, s0);
            s0 = wmma_bf16(aq1, b01, s0);
            s1 = wmma_bf16(aq0, b10, s1);
            s1 = wmma_bf16(aq1, b11, s1);
        }

        // ---- online softmax update (fp32, DPP row reductions) ----
        #pragma unroll
        for (int j = 0; j < 8; ++j) {
            float v0 = s0[j] * 0.125f;   // 1/sqrt(64)
            float v1 = s1[j] * 0.125f;
            float mx    = red_max16(fmaxf(v0, v1));
            float mnew  = fmaxf(mrow[j], mx);
            float alpha = __expf(mrow[j] - mnew);
            float p0 = __expf(v0 - mnew);
            float p1 = __expf(v1 - mnew);
            float ps = red_sum16(p0 + p1);
            lrow[j] = lrow[j] * alpha + ps;
            mrow[j] = mnew;
            o[0][j] *= alpha; o[1][j] *= alpha;
            o[2][j] *= alpha; o[3][j] *= alpha;
            // P into per-wave LDS (C layout rows -> [row][key])
            Pl[w][j + hi * 8][l15]      = (bf16)p0;
            Pl[w][j + hi * 8][16 + l15] = (bf16)p1;
        }

        // ---- O += P @ V  (V block == K block; B from transposed copy) ----
        {
            const bf16* pr = &Pl[w][l15][0];
            bf16x16 ap = make_op(pr + koff, pr + koff + 16);
            #pragma unroll
            for (int nt = 0; nt < 4; ++nt) {
                const bf16* vr = &Kt[nt * 16 + l15][0];   // B: N=dim, K=key
                bf16x16 bv = make_op(vr + koff, vr + koff + 16);
                o[nt] = wmma_bf16(ap, bv, o[nt]);
            }
        }
        __syncthreads();
    }

    // ---- epilogue: normalize and scatter into zc layout (bf16) ----
    #pragma unroll
    for (int nt = 0; nt < 4; ++nt) {
        #pragma unroll
        for (int j = 0; j < 8; ++j) {
            int   n   = q0 + j + hi * 8;
            int   d   = nt * 16 + l15;
            float val = o[nt][j] / lrow[j];
            long  row = (long)h * 64 + (n >> 5);
            long  col = (long)(n & 31) * 64 + d;
            ZC[row * NTOK + col] = (bf16)val;
        }
    }
}

// ---------------------------------------------------------------------------
// x = tokens + positional_enc  (faithful: p = 10000 XOR ((2e)//768))
// ---------------------------------------------------------------------------
__global__ __launch_bounds__(256)
void embed_kernel(const float* __restrict__ tok,
                  float* __restrict__ xF, bf16* __restrict__ xB)
{
    long i = (long)blockIdx.x * 256 + threadIdx.x;   // over NTOK*EDIM
    int  e = (int)(i % EDIM);
    long n = i / EDIM;
    int  s = (int)(n % SEQ);
    float p   = (e < EDIM / 2) ? 10000.0f : 10001.0f;  // 10000 ^ 1 = 10001
    float ang = (float)s / p;
    float pe  = (e & 1) ? cosf(ang) : sinf(ang);
    float x   = tok[i] + pe;
    xF[i] = x;
    xB[i] = (bf16)x;
}

// fp32 -> bf16 weight conversion
__global__ __launch_bounds__(256)
void cvt_bf16(const float* __restrict__ in, bf16* __restrict__ out, long n)
{
    long i = (long)blockIdx.x * 256 + threadIdx.x;
    if (i < n) out[i] = (bf16)in[i];
}

// ---------------------------------------------------------------------------
// out = LayerNorm(X + Y (+bias)) * g + b ; writes fp32 and bf16 copies.
// Y is read flat as [n*768+e] — this *is* the reference's mh reshape, since
// the [768,2048] GEMM output and the [2048,768] token view share memory.
// ---------------------------------------------------------------------------
__global__ __launch_bounds__(256)
void ln_kernel(const float* __restrict__ X, const float* __restrict__ Y,
               const float* __restrict__ bias,
               const float* __restrict__ g, const float* __restrict__ bb,
               float* __restrict__ outF, bf16* __restrict__ outB)
{
    __shared__ float red1[8], red2[8];
    const int n = blockIdx.x, t = threadIdx.x, lane = t & 31, w = t >> 5;

    float v[3];
    #pragma unroll
    for (int i = 0; i < 3; ++i) {
        int e = t + i * 256;
        float y = Y[(long)n * EDIM + e];
        if (bias) y += bias[e];
        v[i] = X[(long)n * EDIM + e] + y;
    }
    float s = v[0] + v[1] + v[2];
    #pragma unroll
    for (int m = 1; m < 32; m <<= 1) s += __shfl_xor(s, m);
    if (lane == 0) red1[w] = s;
    __syncthreads();
    float mu = 0.0f;
    #pragma unroll
    for (int i = 0; i < 8; ++i) mu += red1[i];
    mu *= (1.0f / EDIM);

    float s2 = 0.0f;
    #pragma unroll
    for (int i = 0; i < 3; ++i) { float d = v[i] - mu; s2 += d * d; }
    #pragma unroll
    for (int m = 1; m < 32; m <<= 1) s2 += __shfl_xor(s2, m);
    if (lane == 0) red2[w] = s2;
    __syncthreads();
    float var = 0.0f;
    #pragma unroll
    for (int i = 0; i < 8; ++i) var += red2[i];
    var *= (1.0f / EDIM);

    float inv = rsqrtf(var + 1e-5f);
    #pragma unroll
    for (int i = 0; i < 3; ++i) {
        int e = t + i * 256;
        float o = (v[i] - mu) * inv * g[e] + bb[e];
        outF[(long)n * EDIM + e] = o;
        outB[(long)n * EDIM + e] = (bf16)o;
    }
}

// ---------------------------------------------------------------------------
extern "C" void kernel_launch(void* const* d_in, const int* in_sizes, int n_in,
                              void* d_out, int out_size, void* d_ws, size_t ws_size,
                              hipStream_t stream)
{
    const float* tokens = (const float*)d_in[0];
    const float* wq     = (const float*)d_in[1];
    const float* wo     = (const float*)d_in[2];
    const float* ln1_g  = (const float*)d_in[3];
    const float* ln1_b  = (const float*)d_in[4];
    const float* ffn_w  = (const float*)d_in[5];
    const float* ffn_b  = (const float*)d_in[6];
    const float* ln2_g  = (const float*)d_in[7];
    const float* ln2_b  = (const float*)d_in[8];

    const long NE = (long)NTOK * EDIM;          // 1,572,864
    const long WE = (long)EDIM * EDIM;          // 589,824 (== H*E*D)

    char* p = (char*)d_ws;
    auto carve = [&](size_t bytes) {
        char* r = p;
        p += (bytes + 255) & ~(size_t)255;
        return r;
    };
    float* xA    = (float*)carve(NE * 4);
    float* xB2   = (float*)carve(NE * 4);
    bf16*  x_bf  = (bf16*) carve(NE * 2);
    bf16*  q_bf  = (bf16*) carve((long)NHEAD * NTOK * DDIM * 2);
    bf16*  zc_bf = (bf16*) carve(NE * 2);
    float* mh    = (float*)carve(NE * 4);       // reused for FFN output
    float* h1    = (float*)carve(NE * 4);
    bf16*  h1_bf = (bf16*) carve(NE * 2);
    bf16*  wq_bf = (bf16*) carve(WE * 2);
    bf16*  wo_bf = (bf16*) carve(WE * 2);
    bf16*  fw_bf = (bf16*) carve(WE * 2);

    // x0 = pe + tokens
    embed_kernel<<<(int)(NE / 256), 256, 0, stream>>>(tokens, xA, x_bf);

    float* xin  = xA;
    float* xout = xB2;

    // faithful: encoder(layer 0) once, then layers 0..11 again -> 13 calls
    for (int it = 0; it < 13; ++it) {
        const int l = (it == 0) ? 0 : (it - 1);

        cvt_bf16<<<(int)((WE + 255) / 256), 256, 0, stream>>>(wq    + (long)l * WE, wq_bf, WE);
        cvt_bf16<<<(int)((WE + 255) / 256), 256, 0, stream>>>(wo    + (long)l * WE, wo_bf, WE);
        cvt_bf16<<<(int)((WE + 255) / 256), 256, 0, stream>>>(ffn_w + (long)l * WE, fw_bf, WE);

        // q[h] = X @ Wq[h]   (batched over 12 heads, bf16 out)
        gemm_wmma<false, false><<<dim3(1, 16, NHEAD), 256, 0, stream>>>(
            x_bf, wq_bf, q_bf, nullptr,
            NTOK, DDIM, EDIM, EDIM, DDIM, DDIM,
            0L, (long)EDIM * DDIM, (long)NTOK * DDIM);

        // fused attention -> zc (reference's z.reshape(768,2048) layout)
        flash_attn<<<dim3(16, NHEAD), 256, 0, stream>>>(q_bf, zc_bf);

        // mh = wo @ zc   [768,768]@[768,2048] -> fp32, row-major (= flat mh)
        gemm_wmma<false, true><<<dim3(32, 6, 1), 256, 0, stream>>>(
            wo_bf, zc_bf, nullptr, mh,
            EDIM, NTOK, EDIM, EDIM, NTOK, NTOK, 0L, 0L, 0L);

        // l1 = LN(x + mh)
        ln_kernel<<<NTOK, 256, 0, stream>>>(
            xin, mh, nullptr, ln1_g + (long)l * EDIM, ln1_b + (long)l * EDIM,
            h1, h1_bf);

        // ffn = l1 @ fw^T   (B in [N][K] torch-Linear layout), fp32 out
        gemm_wmma<true, true><<<dim3(12, 16, 1), 256, 0, stream>>>(
            h1_bf, fw_bf, nullptr, mh,
            NTOK, EDIM, EDIM, EDIM, EDIM, EDIM, 0L, 0L, 0L);

        // x_next = LN(l1 + ffn + fb) ; also refresh x_bf for next layer
        ln_kernel<<<NTOK, 256, 0, stream>>>(
            h1, mh, ffn_b + (long)l * EDIM,
            ln2_g + (long)l * EDIM, ln2_b + (long)l * EDIM,
            xout, x_bf);

        float* tmp = xin; xin = xout; xout = tmp;
    }

    (void)hipMemcpyAsync(d_out, xin, NE * sizeof(float),
                         hipMemcpyDeviceToDevice, stream);
}
